// JanossyPooling_31361851195593
// MI455X (gfx1250) — matrix operand order repaired and checked by reading, not compiled, not run on hardware
//
#include <hip/hip_runtime.h>

typedef __attribute__((ext_vector_type(16))) __bf16 v16bf;
typedef __attribute__((ext_vector_type(8)))  __bf16 v8bf;
typedef __attribute__((ext_vector_type(4)))  __bf16 v4bf;
typedef __attribute__((ext_vector_type(8)))  float  v8f;

#define ACT_STRIDE 136   // 128 + 8 halfs pad (bank-conflict avoidance)
#define Y_STRIDE   132   // 128 + 4 floats pad

#ifndef __has_builtin
#define __has_builtin(x) 0
#endif
#if __has_builtin(__builtin_amdgcn_global_load_async_to_lds_b128) && \
    __has_builtin(__builtin_amdgcn_s_wait_asynccnt)
#define USE_ASYNC_GATHER 1
typedef __attribute__((ext_vector_type(4))) int i4vec;
typedef __attribute__((address_space(1))) i4vec as1_i4;   // global int4*
typedef __attribute__((address_space(3))) i4vec as3_i4;   // LDS int4*
#else
#define USE_ASYNC_GATHER 0
#endif

// ---------------------------------------------------------------------------
// One-pass f32 -> bf16 conversion of the atom embedding table (N1 x 128).
// Halves all downstream gather traffic and lets the LDS fill be a raw copy.
// ---------------------------------------------------------------------------
__global__ __launch_bounds__(256) void cvt_atoms_kernel(const float* __restrict__ src,
                                                        __bf16* __restrict__ dst, int n4) {
  int t = blockIdx.x * 256 + threadIdx.x;
  if (t >= n4) return;
  float4 v = ((const float4*)src)[t];
  v4bf o;
  o[0] = (__bf16)v.x; o[1] = (__bf16)v.y; o[2] = (__bf16)v.z; o[3] = (__bf16)v.w;
  ((v4bf*)dst)[t] = o;
}

// ---------------------------------------------------------------------------
// Pack f32 weight matrix [din,128] (row-major, K-major) into bf16 WMMA
// B-fragment layout for v_wmma_f32_16x16x32_bf16:
//   fragment f = kc*8 + ctile ; within fragment: lane L (0..31), element e (0..15)
//   K = kc*32 + (L>=16 ? 8 : 0) + (e & 7) + (e>=8 ? 16 : 0) ;  N = ctile*16 + (L & 15)
// Stored so each lane's 16 halfs are contiguous (32B) -> 2x global_load_b128.
// ---------------------------------------------------------------------------
__global__ __launch_bounds__(256) void pack_weight_kernel(const float* __restrict__ W,
                                                          __bf16* __restrict__ dst,
                                                          int din) {
  int tid = blockIdx.x * 256 + threadIdx.x;
  if (tid >= din * 128) return;
  int e  = tid & 15;
  int L  = (tid >> 4) & 31;
  int fc = tid >> 9;          // fragment index = kc*8 + ctile
  int kc = fc >> 3;
  int c  = fc & 7;
  int k  = kc * 32 + ((L >> 4) * 8) + (e & 7) + ((e >> 3) * 16);
  int n  = c * 16 + (L & 15);
  dst[tid] = (__bf16)W[k * 128 + n];
}

// A-fragment: lane L holds row M = L&15; elems 0..7 = K kbase..kbase+7,
// elems 8..15 = K kbase+16..kbase+23, kbase = kc*32 + 8*(L>>4).
__device__ __forceinline__ v16bf load_afrag(const __bf16* p1, const __bf16* p2) {
  v8bf lo = *(const v8bf*)p1;
  v8bf hi = *(const v8bf*)p2;
  v16bf a;
#pragma unroll
  for (int i = 0; i < 8; ++i) { a[i] = lo[i]; a[i + 8] = hi[i]; }
  return a;
}

// ---------------------------------------------------------------------------
// One full layer for this wave's 16-col tile over TWO 16-row slabs, fwd (+rev)
// sharing every B fragment: each B fragment feeds up to 4 independent WMMAs.
// REVP: layer-1 reversed-tuple path = member-block-permuted reads of srcR
// (each 8-half fragment group always stays inside one 128-wide member block).
// ---------------------------------------------------------------------------
template<int DINL, int LVL, bool HASR, bool REVP>
__device__ __forceinline__ void layer_gemm(const __bf16* __restrict__ srcF,
                                           const __bf16* __restrict__ srcR,
                                           int stride,
                                           const __bf16* __restrict__ Wp,
                                           int lane, int ctile,
                                           v8f& f0, v8f& f1, v8f& r0, v8f& r1) {
  const int rowh = lane & 15;
  const int khi  = (lane >> 4) * 8;
  const __bf16* aF0 = srcF + rowh * stride;
  const __bf16* aF1 = srcF + (16 + rowh) * stride;
  const __bf16* aR0 = srcR + rowh * stride;
  const __bf16* aR1 = srcR + (16 + rowh) * stride;
#pragma unroll
  for (int kc = 0; kc < DINL / 32; ++kc) {
    v16bf b = *(const v16bf*)(Wp + (((size_t)(kc * 8 + ctile)) * 32 + lane) * 16);
    int kbase = kc * 32 + khi;
    int kF1 = kbase, kF2 = kbase + 16;
    int kR1 = kF1, kR2 = kF2;
    if (REVP) {
      int blk = kbase >> 7;
      int off = kbase & 127;
      kR1 = (LVL - 1 - blk) * 128 + off;
      kR2 = kR1 + 16;
    }
    f0 = __builtin_amdgcn_wmma_f32_16x16x32_bf16(false, load_afrag(aF0 + kF1, aF0 + kF2),
                                                 false, b, (short)0, f0, false, false);
    f1 = __builtin_amdgcn_wmma_f32_16x16x32_bf16(false, load_afrag(aF1 + kF1, aF1 + kF2),
                                                 false, b, (short)0, f1, false, false);
    if (HASR) {
      r0 = __builtin_amdgcn_wmma_f32_16x16x32_bf16(false, load_afrag(aR0 + kR1, aR0 + kR2),
                                                   false, b, (short)0, r0, false, false);
      r1 = __builtin_amdgcn_wmma_f32_16x16x32_bf16(false, load_afrag(aR1 + kR1, aR1 + kR2),
                                                   false, b, (short)0, r1, false, false);
    }
  }
}

// C/D layout: element i of v8f -> row = i + 8*(lane>>4) (+slab*16),
// col = ctile*16 + (lane&15)
__device__ __forceinline__ void store_act(v8f acc, float bias,
                                          __bf16* __restrict__ dst,
                                          int lane, int ctile, int slab) {
  int col   = ctile * 16 + (lane & 15);
  int rbase = slab * 16 + (lane >> 4) * 8;
#pragma unroll
  for (int i = 0; i < 8; ++i) {
    float v = acc[i] + bias;
    v = v > 0.f ? v : 0.f;
    dst[(rbase + i) * ACT_STRIDE + col] = (__bf16)v;
  }
}

// ---------------------------------------------------------------------------
// Janossy MLP level kernel. Block = 256 threads = 8 waves, 32 rows per block
// (two 16-row slabs). Wave w owns output column tile w of every layer.
// Shared-memory pool with lifetime aliasing keeps static LDS <= 51.7 KB.
// ---------------------------------------------------------------------------
template<int LVL, int NHEADS>
__global__ __launch_bounds__(256) void janossy_mlp(
    const __bf16* __restrict__ atomsBf,
    const int*    __restrict__ idx,
    int N,
    const __bf16* __restrict__ W1, const __bf16* __restrict__ W2, const __bf16* __restrict__ W3,
    const float* __restrict__ b1,  const float* __restrict__ b2,  const float* __restrict__ b3,
    const float* __restrict__ hW0, const float* __restrict__ hW1p, const float* __restrict__ hW2p,
    const float* __restrict__ hB0, const float* __restrict__ hB1p, const float* __restrict__ hB2p,
    float* __restrict__ o0, float* __restrict__ o1p, float* __restrict__ o2p) {

  constexpr int    DIN   = 128 * LVL;
  constexpr int    INS   = DIN + 8;                              // input row stride (halfs)
  constexpr size_t SZ_A  = (size_t)32 * ACT_STRIDE * 2;          // 8704 B
  constexpr size_t SZ_Y  = (size_t)32 * Y_STRIDE * 4;            // 16896 B
  constexpr size_t SZ_IN = (size_t)32 * INS * 2;
  constexpr size_t NEED0 = (LVL > 1) ? (2 * SZ_A + SZ_Y) : (SZ_A + SZ_Y);
  constexpr size_t R0    = (SZ_IN > NEED0) ? SZ_IN : NEED0;      // region0 size
  constexpr size_t TOTAL = R0 + ((LVL > 1) ? 2 * SZ_A : SZ_A);

  __shared__ __align__(16) char smem[TOTAL];
  __bf16* sIn = (__bf16*)smem;                       // gather buf (dead after L1)
  __bf16* A0  = (__bf16*)(smem + R0);                // L1 fwd out
  __bf16* A1  = (LVL > 1) ? (__bf16*)(smem + R0 + SZ_A)  // L1 rev out
                          : (__bf16*)smem;               // lvl1: L2 out aliases sIn
  __bf16* A2  = (__bf16*)smem;                       // L2 fwd out (aliases sIn)
  __bf16* A3  = (__bf16*)(smem + SZ_A);              // L2 rev out (aliases sIn)
  float*  sY  = (LVL > 1) ? (float*)(smem + 2 * SZ_A)
                          : (float*)(smem + SZ_A);

  const int tid    = threadIdx.x;
  const int lane   = tid & 31;
  const int wave   = tid >> 5;               // column tile 0..7
  const int rowBlk = blockIdx.x * 32;

  // ---- gather: bf16 atom rows -> LDS (async direct-to-LDS when available) ----
  constexpr int CHUNKS = 32 * LVL * 8;       // 16-half (32B) segments
  for (int ch = tid; ch < CHUNKS; ch += 256) {
    int r    = ch / (LVL * 8);
    int rem  = ch - r * (LVL * 8);
    int m    = rem >> 3;
    int seg  = rem & 7;
    int grow = rowBlk + r;
    int atom;
    if constexpr (LVL == 1) { atom = (grow < N) ? grow : 0; }
    else                    { atom = (grow < N) ? idx[grow * LVL + m] : 0; }
    const __bf16* s = atomsBf + (size_t)atom * 128 + seg * 16;
    __bf16*       d = sIn + r * INS + m * 128 + seg * 16;
#if USE_ASYNC_GATHER
    __builtin_amdgcn_global_load_async_to_lds_b128((as1_i4*)s, (as3_i4*)d, 0, 0);
    __builtin_amdgcn_global_load_async_to_lds_b128((as1_i4*)(s + 8), (as3_i4*)(d + 8), 0, 0);
#else
    *(uint4*)d       = *(const uint4*)s;
    *(uint4*)(d + 8) = *(const uint4*)(s + 8);
#endif
  }
#if USE_ASYNC_GATHER
  __builtin_amdgcn_s_wait_asynccnt(0);
#endif
  __syncthreads();

  // ---- layer 1: sIn (fwd + member-reversed) -> A0 / A1 ----
  {
    float bias = b1[wave * 16 + (lane & 15)];
    v8f f0 = {}, f1 = {}, r0 = {}, r1 = {};
    layer_gemm<DIN, LVL, (LVL > 1), (LVL > 1)>(sIn, sIn, INS, W1, lane, wave, f0, f1, r0, r1);
    store_act(f0, bias, A0, lane, wave, 0);
    store_act(f1, bias, A0, lane, wave, 1);
    if constexpr (LVL > 1) {
      store_act(r0, bias, A1, lane, wave, 0);
      store_act(r1, bias, A1, lane, wave, 1);
    }
  }
  __syncthreads();

  // ---- layer 2: A0/A1 -> A2/A3 (lvl1: A0 -> A1) ----
  {
    float bias = b2[wave * 16 + (lane & 15)];
    v8f f0 = {}, f1 = {}, r0 = {}, r1 = {};
    layer_gemm<128, 1, (LVL > 1), false>(A0, A1, ACT_STRIDE, W2, lane, wave, f0, f1, r0, r1);
    if constexpr (LVL > 1) {
      store_act(f0, bias, A2, lane, wave, 0);
      store_act(f1, bias, A2, lane, wave, 1);
      store_act(r0, bias, A3, lane, wave, 0);
      store_act(r1, bias, A3, lane, wave, 1);
    } else {
      store_act(f0, bias, A1, lane, wave, 0);
      store_act(f1, bias, A1, lane, wave, 1);
    }
  }
  __syncthreads();

  // ---- layer 3: y = relu(fwd) + relu(rev) -> f32 LDS ----
  {
    float bias = b3[wave * 16 + (lane & 15)];
    int col   = wave * 16 + (lane & 15);
    int rbase = (lane >> 4) * 8;
    v8f f0 = {}, f1 = {}, r0 = {}, r1 = {};
    if constexpr (LVL > 1) {
      layer_gemm<128, 1, true, false>(A2, A3, ACT_STRIDE, W3, lane, wave, f0, f1, r0, r1);
#pragma unroll
      for (int i = 0; i < 8; ++i) {
        float a = f0[i] + bias; a = a > 0.f ? a : 0.f;
        float b = r0[i] + bias; b = b > 0.f ? b : 0.f;
        sY[(rbase + i) * Y_STRIDE + col] = a + b;
        float c = f1[i] + bias; c = c > 0.f ? c : 0.f;
        float d = r1[i] + bias; d = d > 0.f ? d : 0.f;
        sY[(16 + rbase + i) * Y_STRIDE + col] = c + d;
      }
    } else {
      layer_gemm<128, 1, false, false>(A1, A1, ACT_STRIDE, W3, lane, wave, f0, f1, r0, r1);
#pragma unroll
      for (int i = 0; i < 8; ++i) {
        float a = f0[i] + bias; a = a > 0.f ? a : 0.f;
        sY[(rbase + i) * Y_STRIDE + col] = a;
        float c = f1[i] + bias; c = c > 0.f ? c : 0.f;
        sY[(16 + rbase + i) * Y_STRIDE + col] = c;
      }
    }
  }
  __syncthreads();

  // ---- heads: y[32x128] @ hW[128] + hb, shuffle-reduced dot per row ----
  {
    const float* hW[3]  = {hW0, hW1p, hW2p};
    const float* hB[3]  = {hB0, hB1p, hB2p};
    float*       out[3] = {o0, o1p, o2p};
    int row  = tid >> 3;                 // 0..31
    int part = tid & 7;                  // 8 partials x 16 elems
    int grow = rowBlk + row;
#pragma unroll
    for (int h = 0; h < NHEADS; ++h) {
      float s = 0.f;
#pragma unroll
      for (int k = 0; k < 16; ++k)
        s += sY[row * Y_STRIDE + part * 16 + k] * hW[h][part * 16 + k];
      s += __shfl_down(s, 4, 8);
      s += __shfl_down(s, 2, 8);
      s += __shfl_down(s, 1, 8);
      if (part == 0 && grow < N) out[h][grow] = s + hB[h][0];
    }
  }
}

// ---------------------------------------------------------------------------
// d_in layout (setup_inputs dict order, params insertion order, (W,b) pairs):
//  0 atom_h | 1 idx2 | 2 idx3 | 3 idx4
//  4..9   seq1 (W1,b1,W2,b2,W3,b3)   | 10..15 head1 sigma/eps/q (W,b each)
// 16..21  seq2                        | 22..25 head2 k/eq
// 26..31  seq3                        | 32..35 head3 k/eq
// 36..41  seq4                        | 42..45 head4 k/eq
// ---------------------------------------------------------------------------
extern "C" void kernel_launch(void* const* d_in, const int* in_sizes, int n_in,
                              void* d_out, int out_size, void* d_ws, size_t ws_size,
                              hipStream_t stream) {
  const float* atom_h = (const float*)d_in[0];
  const int*   idx2   = (const int*)d_in[1];
  const int*   idx3   = (const int*)d_in[2];
  const int*   idx4   = (const int*)d_in[3];

  const int N1 = in_sizes[0] / 128;
  const int N2 = in_sizes[1] / 2;
  const int N3 = in_sizes[2] / 3;
  const int N4 = in_sizes[3] / 4;

  __bf16* ws  = (__bf16*)d_ws;
  size_t  off = 0;
  auto pack = [&](int inIdx, int din) -> __bf16* {
    __bf16* p = ws + off;
    int total = din * 128;
    pack_weight_kernel<<<(total + 255) / 256, 256, 0, stream>>>(
        (const float*)d_in[inIdx], p, din);
    off += (size_t)total;
    return p;
  };

  __bf16 *w1a = pack(4, 128),  *w1b = pack(6, 128),  *w1c = pack(8, 128);
  __bf16 *w2a = pack(16, 256), *w2b = pack(18, 128), *w2c = pack(20, 128);
  __bf16 *w3a = pack(26, 384), *w3b = pack(28, 128), *w3c = pack(30, 128);
  __bf16 *w4a = pack(36, 512), *w4b = pack(38, 128), *w4c = pack(40, 128);

  // bf16 atom table (gathers read this; half the bytes of the f32 table)
  __bf16* atomsBf = ws + off;
  {
    int n4 = in_sizes[0] / 4;
    cvt_atoms_kernel<<<(n4 + 255) / 256, 256, 0, stream>>>(atom_h, atomsBf, n4);
    off += (size_t)in_sizes[0];
  }

  float* out = (float*)d_out;
  float* o1 = out;                         // 3 x N1
  float* o2 = o1 + 3 * (size_t)N1;         // 2 x N2
  float* o3 = o2 + 2 * (size_t)N2;         // 2 x N3
  float* o4 = o3 + 2 * (size_t)N3;         // 2 x N4

  const float* nfp = nullptr;
  float*       nfo = nullptr;

  janossy_mlp<1, 3><<<(N1 + 31) / 32, 256, 0, stream>>>(
      atomsBf, (const int*)nullptr, N1, w1a, w1b, w1c,
      (const float*)d_in[5], (const float*)d_in[7], (const float*)d_in[9],
      (const float*)d_in[10], (const float*)d_in[12], (const float*)d_in[14],
      (const float*)d_in[11], (const float*)d_in[13], (const float*)d_in[15],
      o1, o1 + N1, o1 + 2 * (size_t)N1);

  janossy_mlp<2, 2><<<(N2 + 31) / 32, 256, 0, stream>>>(
      atomsBf, idx2, N2, w2a, w2b, w2c,
      (const float*)d_in[17], (const float*)d_in[19], (const float*)d_in[21],
      (const float*)d_in[22], (const float*)d_in[24], nfp,
      (const float*)d_in[23], (const float*)d_in[25], nfp,
      o2, o2 + N2, nfo);

  janossy_mlp<3, 2><<<(N3 + 31) / 32, 256, 0, stream>>>(
      atomsBf, idx3, N3, w3a, w3b, w3c,
      (const float*)d_in[27], (const float*)d_in[29], (const float*)d_in[31],
      (const float*)d_in[32], (const float*)d_in[34], nfp,
      (const float*)d_in[33], (const float*)d_in[35], nfp,
      o3, o3 + N3, nfo);

  janossy_mlp<4, 2><<<(N4 + 31) / 32, 256, 0, stream>>>(
      atomsBf, idx4, N4, w4a, w4b, w4c,
      (const float*)d_in[37], (const float*)d_in[39], (const float*)d_in[41],
      (const float*)d_in[42], (const float*)d_in[44], nfp,
      (const float*)d_in[43], (const float*)d_in[45], nfp,
      o4, o4 + N4, nfo);
}
